// multiGCNEncoder_15195594293387
// MI455X (gfx1250) — matrix operand order repaired and checked by reading, not compiled, not run on hardware
//
#include <hip/hip_runtime.h>
#include <hip/hip_bf16.h>
#include <stdint.h>

typedef __bf16 bf16_t;
typedef __attribute__((ext_vector_type(16))) __bf16 v16bf;
typedef __attribute__((ext_vector_type(8)))  __bf16 v8bf;
typedef __attribute__((ext_vector_type(8)))  float  v8f;

#define NPAT  8192
#define DFIN  128
#define SLOPE 0.25f

// ---------------- elementwise helpers ----------------

__global__ void k_f32_to_bf16(const float* __restrict__ src, bf16_t* __restrict__ dst, int n) {
  int i = blockIdx.x * blockDim.x + threadIdx.x;
  if (i < n) dst[i] = (bf16_t)src[i];
}

// Wt[n*K + k] = bf16(W[k*Nw + n])  (transpose to N-major for B-fragment loads)
__global__ void k_transpose_to_bf16(const float* __restrict__ W, bf16_t* __restrict__ Wt,
                                    int K, int Nw) {
  int i = blockIdx.x * blockDim.x + threadIdx.x;
  if (i >= K * Nw) return;
  int k = i / Nw, n = i - k * Nw;
  Wt[(size_t)n * K + k] = (bf16_t)W[i];
}

__global__ void k_zero_f32(float* __restrict__ p, int n) {
  int i = blockIdx.x * blockDim.x + threadIdx.x;
  if (i < n) p[i] = 0.0f;
}

// ---------------- WMMA GEMM: C[MxN] = A[MxK](bf16,row-major) * Bt[NxK]^T ----------------
// One wave computes a 64x64 tile as 4x4 grid of v_wmma_f32_16x16x32_bf16.
__global__ __launch_bounds__(128) void k_gemm_wmma_bf16(
    const bf16_t* __restrict__ A, const bf16_t* __restrict__ Bt,
    float* __restrict__ C, int M, int N, int K) {
  const int lane    = threadIdx.x & 31;
  const int wave    = threadIdx.x >> 5;
  const int nTilesN = N >> 6;
  const int nTiles  = (M >> 6) * nTilesN;
  const int tile    = blockIdx.x * 4 + wave;
  if (tile >= nTiles) return;           // uniform per wave: EXEC stays all-1s for WMMA
  const int tm   = (tile / nTilesN) << 6;
  const int tn   = (tile % nTilesN) << 6;
  const int half = lane >> 4;           // K sub-range selector (this code's contiguous order: K = half*16 .. +15)
  const int l16  = lane & 15;

  v8f zero = {0.f,0.f,0.f,0.f,0.f,0.f,0.f,0.f};
  v8f acc[4][4];
#pragma unroll
  for (int mi = 0; mi < 4; ++mi)
#pragma unroll
    for (int ni = 0; ni < 4; ++ni) acc[mi][ni] = zero;

  for (int k0 = 0; k0 < K; k0 += 32) {
    v16bf a[4], b[4];
    // A 16x32 fragment: lanes 0-15 rows M=l16 hold K 0..15; lanes 16-31 hold K 16..31
#pragma unroll
    for (int mi = 0; mi < 4; ++mi) {
      const bf16_t* p = A + (size_t)(tm + mi * 16 + l16) * K + k0 + half * 16;
      v8bf lo = *(const v8bf*)p;            // K = k0 + half*16 .. +7
      v8bf hi = *(const v8bf*)(p + 8);      // K = k0 + half*16 + 8 .. +7
      v16bf f;
#pragma unroll
      for (int t = 0; t < 8; ++t) { f[t] = lo[t]; f[8 + t] = hi[t]; }
      a[mi] = f;
    }
    // B 32x16 fragment: lanes 0-15 hold K=k0..k0+15 of column N=l16; lanes 16-31 hold K=k0+16..k0+31
#pragma unroll
    for (int ni = 0; ni < 4; ++ni) {
      const bf16_t* q = Bt + (size_t)(tn + ni * 16 + l16) * K + k0 + (half << 4);
      b[ni] = *(const v16bf*)q;
    }
#pragma unroll
    for (int mi = 0; mi < 4; ++mi)
#pragma unroll
      for (int ni = 0; ni < 4; ++ni)
        acc[mi][ni] = __builtin_amdgcn_wmma_f32_16x16x32_bf16(
            false, a[mi], false, b[ni], (short)0, acc[mi][ni], false, false);
  }

  // C/D layout: VGPR j -> row (half*8 + j), col l16
#pragma unroll
  for (int mi = 0; mi < 4; ++mi)
#pragma unroll
    for (int ni = 0; ni < 4; ++ni) {
      float* cp = C + (size_t)(tm + mi * 16 + half * 8) * N + tn + ni * 16 + l16;
#pragma unroll
      for (int j = 0; j < 8; ++j) cp[(size_t)j * N] = acc[mi][ni][j];
    }
}

// ---------------- SpMM scatter: out[rows[e]] += vals[e] * dense[cols[e]] ----------------
__global__ void k_spmm_scatter(const int* __restrict__ rows, const int* __restrict__ cols,
                               const float* __restrict__ vals, const float* __restrict__ dense,
                               float* __restrict__ out, int E, int ld) {
  int gw   = (int)((blockIdx.x * blockDim.x + threadIdx.x) >> 5);
  int lane = threadIdx.x & 31;
  if (gw >= E) return;
  int   r = rows[gw], c = cols[gw];
  float v = vals[gw];
  const float* src = dense + (size_t)c * ld;
  float*       dst = out   + (size_t)r * ld;
  for (int j = lane; j < ld; j += 32)
    atomicAdd(dst + j, v * src[j]);
}

// ---------------- bias + leaky_relu (+convert / +accumulate) ----------------
__global__ void k_bias_lrelu_bf16(const float* __restrict__ H, const float* __restrict__ bias,
                                  bf16_t* __restrict__ out, int ld, int n) {
  int i = blockIdx.x * blockDim.x + threadIdx.x;
  if (i >= n) return;
  float h = H[i] + bias[i % ld];
  h = h > 0.f ? h : SLOPE * h;
  out[i] = (bf16_t)h;
}

__global__ void k_bias_lrelu_accum(const float* __restrict__ H, const float* __restrict__ bias,
                                   float* __restrict__ out, int ld, int n, int first) {
  int i = blockIdx.x * blockDim.x + threadIdx.x;
  if (i >= n) return;
  float h = H[i] + bias[i % ld];
  h = h > 0.f ? h : SLOPE * h;
  h *= 0.25f;
  if (first) out[i] = h; else out[i] += h;
}

// ---------------- driver ----------------

static inline size_t al256(size_t x) { return (x + 255) & ~(size_t)255; }

extern "C" void kernel_launch(void* const* d_in, const int* in_sizes, int n_in,
                              void* d_out, int out_size, void* d_ws, size_t ws_size,
                              hipStream_t stream) {
  (void)n_in; (void)out_size; (void)ws_size;
  const int E = in_sizes[4];

  const int din_arr[4] = {4096, 4096, 4096, 1024};
  const int dh_arr[4]  = {512,  512,  512,  256};

  // workspace layout (sized for the largest branch)
  char*  ws  = (char*)d_ws;
  size_t off = 0;
  bf16_t* Xb  = (bf16_t*)(ws + off); off += al256((size_t)NPAT * 4096 * sizeof(bf16_t));
  bf16_t* W1t = (bf16_t*)(ws + off); off += al256((size_t)4096 * 512 * sizeof(bf16_t));
  float*  XW  = (float*) (ws + off); off += al256((size_t)NPAT * 512 * sizeof(float));
  float*  H1  = (float*) (ws + off); off += al256((size_t)NPAT * 512 * sizeof(float));
  bf16_t* H1b = (bf16_t*)(ws + off); off += al256((size_t)NPAT * 512 * sizeof(bf16_t));
  bf16_t* W2t = (bf16_t*)(ws + off); off += al256((size_t)512 * DFIN * sizeof(bf16_t));
  float*  HW  = (float*) (ws + off); off += al256((size_t)NPAT * DFIN * sizeof(float));
  float*  H2  = (float*) (ws + off); off += al256((size_t)NPAT * DFIN * sizeof(float));

  float* out = (float*)d_out;

  for (int g = 0; g < 4; ++g) {
    const int din = din_arr[g];
    const int dh  = dh_arr[g];
    const float* x    = (const float*)d_in[g];
    const int*   rows = (const int*)  d_in[4 + 3 * g];
    const int*   cols = (const int*)  d_in[5 + 3 * g];
    const float* vals = (const float*)d_in[6 + 3 * g];
    const float* w1   = (const float*)d_in[16 + 4 * g];
    const float* b1   = (const float*)d_in[17 + 4 * g];
    const float* w2   = (const float*)d_in[18 + 4 * g];
    const float* b2   = (const float*)d_in[19 + 4 * g];

    // 1) X -> bf16
    {
      int n = NPAT * din;
      k_f32_to_bf16<<<(n + 255) / 256, 256, 0, stream>>>(x, Xb, n);
    }
    // 2) W1 -> bf16, N-major
    {
      int n = din * dh;
      k_transpose_to_bf16<<<(n + 255) / 256, 256, 0, stream>>>(w1, W1t, din, dh);
    }
    // 3) XW = Xb @ W1
    {
      int tiles = (NPAT >> 6) * (dh >> 6);
      k_gemm_wmma_bf16<<<(tiles + 3) / 4, 128, 0, stream>>>(Xb, W1t, XW, NPAT, dh, din);
    }
    // 4-5) H1 = spmm(A, XW)
    {
      int n = NPAT * dh;
      k_zero_f32<<<(n + 255) / 256, 256, 0, stream>>>(H1, n);
      k_spmm_scatter<<<(E + 7) / 8, 256, 0, stream>>>(rows, cols, vals, XW, H1, E, dh);
    }
    // 6) H1b = bf16(lrelu(H1 + b1))
    {
      int n = NPAT * dh;
      k_bias_lrelu_bf16<<<(n + 255) / 256, 256, 0, stream>>>(H1, b1, H1b, dh, n);
    }
    // 7) W2 -> bf16, N-major
    {
      int n = dh * DFIN;
      k_transpose_to_bf16<<<(n + 255) / 256, 256, 0, stream>>>(w2, W2t, dh, DFIN);
    }
    // 8) HW = H1b @ W2
    {
      int tiles = (NPAT >> 6) * (DFIN >> 6);
      k_gemm_wmma_bf16<<<(tiles + 3) / 4, 128, 0, stream>>>(H1b, W2t, HW, NPAT, DFIN, dh);
    }
    // 9-10) H2 = spmm(A, HW)
    {
      int n = NPAT * DFIN;
      k_zero_f32<<<(n + 255) / 256, 256, 0, stream>>>(H2, n);
      k_spmm_scatter<<<(E + 7) / 8, 256, 0, stream>>>(rows, cols, vals, HW, H2, E, DFIN);
    }
    // 11) out (+)= 0.25 * lrelu(H2 + b2)
    {
      int n = NPAT * DFIN;
      k_bias_lrelu_accum<<<(n + 255) / 256, 256, 0, stream>>>(H2, b2, out, DFIN, n, g == 0);
    }
  }
}
